// SelfAttention_71279277244548
// MI455X (gfx1250) — matrix operand order repaired and checked by reading, not compiled, not run on hardware
//
#include <hip/hip_runtime.h>
#include <hip/hip_bf16.h>

// ---------------------------------------------------------------------------
// Self-attention, B=4, S=4096, D=128, f32 in/out.
//   Kernel 1 (qkv_project): Q/K/V projections via v_wmma_f32_16x16x32_f16.
//     W staged per-block in LDS transposed -> each B-fragment is one
//     contiguous 32-byte LDS read. Q,K row-major f16; V transposed
//     (Vt[b][d][s]) f16 in d_ws (12 MB total).
//   Kernel 2 (flash_attn): flash attention; 8 waves/block, 16 q-rows/wave.
//     K/V streamed in 32-key chunks, double-buffered in LDS (one copy per
//     block; global loads of chunk t+1 overlap the 16 WMMAs of chunk t).
//     B-fragments are batch-preloaded into registers before each WMMA chain
//     so the ds waits amortize instead of stalling every v_wmma.
// ---------------------------------------------------------------------------

typedef __attribute__((ext_vector_type(16))) _Float16     v16h;
typedef __attribute__((ext_vector_type(8)))  float        v8f;
typedef __attribute__((ext_vector_type(4)))  unsigned int v4u;   // 16 bytes

#define BATCH 4
#define SEQ   4096
#define DIM   128
#define NKT   (SEQ / 32)

__device__ __forceinline__ v8f wmma_f16(v16h a, v16h b, v8f c) {
    // D = A(16x32) * B(32x16) + C(16x16 f32)
    return __builtin_amdgcn_wmma_f32_16x16x32_f16(
        false, a, false, b, (short)0, c, false, false);
}

// A-fragment (16x32 f16) K-column for pair p (0..7), lane-half hi (0/1):
// lanes 0-15 hold K in {0..15}, lanes 16-31 hold K in {16..31}.
__device__ __forceinline__ int a_kcol(int p, int hi) {
    return 2 * p + 16 * hi;
}

// ---------------------------------------------------------------------------
// Kernel 1: QKV projection. grid = BATCH * (SEQ/128) = 128 blocks, 256 thr.
// ---------------------------------------------------------------------------
__global__ __launch_bounds__(256)
void qkv_project_kernel(const float* __restrict__ X,
                        const float* __restrict__ Wq,
                        const float* __restrict__ Wk,
                        const float* __restrict__ Wv,
                        _Float16* __restrict__ Qh,
                        _Float16* __restrict__ Kh,
                        _Float16* __restrict__ Vt) {
    __shared__ __align__(32) _Float16 Wt[DIM * DIM];     // 32 KB: Wt[n*128 + k]

    const int tid  = threadIdx.x;
    const int lane = tid & 31;
    const int wave = tid >> 5;
    const int lo = lane & 15;
    const int hi = lane >> 4;
    const int b = blockIdx.x >> 5;                       // / 32
    const int sBase = ((blockIdx.x & 31) << 7) + wave * 16;

    // Load X tile (16 rows x 128) as 4 A-fragments (f32 -> f16).
    v16h aX[4];
    const float* xrow = X + (size_t)(b * SEQ + sBase + lo) * DIM;
#pragma unroll
    for (int kc = 0; kc < 4; ++kc) {
#pragma unroll
        for (int p = 0; p < 8; ++p) {
            const int kcol = kc * 32 + a_kcol(p, hi);
            aX[kc][2 * p]     = (_Float16)xrow[kcol];
            aX[kc][2 * p + 1] = (_Float16)xrow[kcol + 1];
        }
    }

    const float* Ws[3] = {Wq, Wk, Wv};
#pragma unroll
    for (int w = 0; w < 3; ++w) {                        // w is compile-time now
        // Stage W transposed in LDS: Wt[n][k] = (f16) W[k][n].
        __syncthreads();
        const float* W = Ws[w];
        for (int i = tid; i < DIM * DIM; i += 256) {
            const int k = i >> 7, n = i & 127;
            Wt[n * DIM + k] = (_Float16)W[i];
        }
        __syncthreads();

        for (int nt = 0; nt < 8; ++nt) {                 // 8 tiles of 16 cols
            // Preload all 4 B-fragments, then run the WMMA chain.
            v16h bW[4];
#pragma unroll
            for (int kc = 0; kc < 4; ++kc)
                bW[kc] = *(const v16h*)(Wt + (nt * 16 + lo) * DIM + kc * 32 + hi * 16);
            v8f acc = {};
#pragma unroll
            for (int kc = 0; kc < 4; ++kc)
                acc = wmma_f16(aX[kc], bW[kc], acc);
#pragma unroll
            for (int r = 0; r < 8; ++r) {                // C layout: row r+8*hi
                const int row = sBase + r + 8 * hi;
                const int col = nt * 16 + lo;
                if (w == 0)
                    Qh[((size_t)b * SEQ + row) * DIM + col] = (_Float16)acc[r];
                else if (w == 1)
                    Kh[((size_t)b * SEQ + row) * DIM + col] = (_Float16)acc[r];
                else
                    Vt[((size_t)b * DIM + col) * SEQ + row] = (_Float16)acc[r];
            }
        }
    }
}

// ---------------------------------------------------------------------------
// Kernel 2: flash attention with double-buffered LDS K/V staging.
// grid = 128 blocks, 256 thr (8 waves), 16 query rows per wave.
// ---------------------------------------------------------------------------
__global__ __launch_bounds__(256)
void flash_attn_kernel(const _Float16* __restrict__ Qh,
                       const _Float16* __restrict__ Kh,
                       const _Float16* __restrict__ Vt,
                       float* __restrict__ out) {
    __shared__ __align__(32) _Float16 Kl[2][32 * DIM];   // 2 x 8 KB (32 keys x 128)
    __shared__ __align__(32) _Float16 Vl[2][DIM * 32];   // 2 x 8 KB (128 dims x 32 keys)
    __shared__ __align__(32) _Float16 Pl[8][16 * 32];    // per-wave P staging, 8 KB

    const int tid  = threadIdx.x;
    const int lane = tid & 31;
    const int wave = tid >> 5;
    const int lo = lane & 15;
    const int hi = lane >> 4;
    const int b = blockIdx.x >> 5;
    const int qBase = ((blockIdx.x & 31) << 7) + wave * 16;

    // --- chunk copy helpers (per-thread 2x16B for K, 2x16B for V) ----------
    const _Float16* Kbase = Kh + (size_t)b * SEQ * DIM;
    const _Float16* Vbase = Vt + ((size_t)b * DIM + (tid >> 1)) * SEQ;
    auto loadChunk = [&](int kt, v4u* kr, v4u* vr) {
        const v4u* ks = (const v4u*)(Kbase + (size_t)kt * 32 * DIM);
        kr[0] = ks[tid * 2];
        kr[1] = ks[tid * 2 + 1];
        const v4u* vs = (const v4u*)(Vbase + kt * 32) + (tid & 1) * 2;
        vr[0] = vs[0];
        vr[1] = vs[1];
    };
    auto storeChunk = [&](int buf, const v4u* kr, const v4u* vr) {
        v4u* kd = (v4u*)Kl[buf];
        kd[tid * 2]     = kr[0];
        kd[tid * 2 + 1] = kr[1];
        v4u* vd = (v4u*)(Vl[buf] + (tid >> 1) * 32) + (tid & 1) * 2;
        vd[0] = vr[0];
        vd[1] = vr[1];
    };

    // --- Q tile as 4 A-fragments -------------------------------------------
    v16h aQ[4];
    const _Float16* qrow = Qh + (size_t)(b * SEQ + qBase + lo) * DIM;
#pragma unroll
    for (int kc = 0; kc < 4; ++kc) {
#pragma unroll
        for (int p = 0; p < 8; ++p) {
            const int kcol = kc * 32 + a_kcol(p, hi);
            aQ[kc][2 * p]     = qrow[kcol];
            aQ[kc][2 * p + 1] = qrow[kcol + 1];
        }
    }

    v8f O[8];
    float m_i[8], l_i[8];
#pragma unroll
    for (int nt = 0; nt < 8; ++nt) O[nt] = (v8f){};
#pragma unroll
    for (int r = 0; r < 8; ++r) { m_i[r] = -__builtin_inff(); l_i[r] = 0.0f; }

    const float scale = 0.0883883476483184f;             // 1/sqrt(128)

    // Prologue: stage chunk 0.
    v4u kr[2], vr[2];
    loadChunk(0, kr, vr);
    storeChunk(0, kr, vr);
    __syncthreads();

    for (int kt = 0; kt < NKT; ++kt) {
        const int buf = kt & 1;
        const bool hasNext = (kt + 1 < NKT);
        if (hasNext) loadChunk(kt + 1, kr, vr);          // overlaps WMMA below

        // Preload all 8 K B-fragments, then the 8-WMMA S chain.
        v16h bK[8];
#pragma unroll
        for (int kc = 0; kc < 4; ++kc) {
            const _Float16* k0 = Kl[buf] + lo * DIM + kc * 32 + hi * 16;
            bK[kc]     = *(const v16h*)k0;
            bK[kc + 4] = *(const v16h*)(k0 + 16 * DIM);
        }
        v8f s0 = {}, s1 = {};
#pragma unroll
        for (int kc = 0; kc < 4; ++kc) {
            s0 = wmma_f16(aQ[kc], bK[kc], s0);
            s1 = wmma_f16(aQ[kc], bK[kc + 4], s1);
        }

        // Online softmax (C-layout rows live in 16-lane halves).
        float corr[8];
#pragma unroll
        for (int r = 0; r < 8; ++r) {
            const float a0 = s0[r] * scale;
            const float a1 = s1[r] * scale;
            float mc = fmaxf(a0, a1);
#pragma unroll
            for (int mask = 1; mask < 16; mask <<= 1)
                mc = fmaxf(mc, __shfl_xor(mc, mask, 32));
            const float mnew = fmaxf(m_i[r], mc);
            const float c  = __expf(m_i[r] - mnew);
            const float p0 = __expf(a0 - mnew);
            const float p1 = __expf(a1 - mnew);
            float rs = p0 + p1;
#pragma unroll
            for (int mask = 1; mask < 16; mask <<= 1)
                rs += __shfl_xor(rs, mask, 32);
            l_i[r] = l_i[r] * c + rs;
            m_i[r] = mnew;
            corr[r] = c;
            // Stage P (C layout -> row-major 16x32) for A-fragment reload.
            Pl[wave][(r + 8 * hi) * 32 + lo]      = (_Float16)p0;
            Pl[wave][(r + 8 * hi) * 32 + 16 + lo] = (_Float16)p1;
        }
#pragma unroll
        for (int nt = 0; nt < 8; ++nt)
#pragma unroll
            for (int r = 0; r < 8; ++r) O[nt][r] *= corr[r];

        __asm__ volatile("" ::: "memory");               // order LDS store->load
        v16h aP;
#pragma unroll
        for (int p = 0; p < 8; ++p) {
            const int kcol = a_kcol(p, hi);
            aP[2 * p]     = Pl[wave][lo * 32 + kcol];
            aP[2 * p + 1] = Pl[wave][lo * 32 + kcol + 1];
        }

        // Preload all 8 V B-fragments, then the 8-WMMA O chain.
        v16h bV[8];
#pragma unroll
        for (int nt = 0; nt < 8; ++nt)
            bV[nt] = *(const v16h*)(Vl[buf] + (nt * 16 + lo) * 32 + hi * 16);
#pragma unroll
        for (int nt = 0; nt < 8; ++nt)
            O[nt] = wmma_f16(aP, bV[nt], O[nt]);

        __syncthreads();                                 // all waves done with buf^1
        if (hasNext) storeChunk(buf ^ 1, kr, vr);
        __syncthreads();                                 // staged data visible
    }

    // Normalize and store f32 output.
#pragma unroll
    for (int r = 0; r < 8; ++r) {
        const float inv = 1.0f / l_i[r];
        const int row = qBase + r + 8 * hi;
#pragma unroll
        for (int nt = 0; nt < 8; ++nt)
            out[((size_t)b * SEQ + row) * DIM + nt * 16 + lo] = O[nt][r] * inv;
    }
}

// ---------------------------------------------------------------------------
extern "C" void kernel_launch(void* const* d_in, const int* in_sizes, int n_in,
                              void* d_out, int out_size, void* d_ws, size_t ws_size,
                              hipStream_t stream) {
    const float* X  = (const float*)d_in[0];
    const float* Wq = (const float*)d_in[1];
    const float* Wk = (const float*)d_in[2];
    const float* Wv = (const float*)d_in[3];
    float* out = (float*)d_out;

    // Workspace: Q (4MB) | K (4MB) | Vt (4MB), all f16.
    const size_t elems = (size_t)BATCH * SEQ * DIM;
    _Float16* Qh = (_Float16*)d_ws;
    _Float16* Kh = Qh + elems;
    _Float16* Vt = Kh + elems;

    const int blocks = BATCH * (SEQ / 128);              // 128 blocks, 8 waves ea.
    qkv_project_kernel<<<blocks, 256, 0, stream>>>(X, Wq, Wk, Wv, Qh, Kh, Vt);
    flash_attn_kernel<<<blocks, 256, 0, stream>>>(Qh, Kh, Vt, out);
}
